// MixOfExperts_17386027615047
// MI455X (gfx1250) — matrix operand order repaired and checked by reading, not compiled, not run on hardware
//
#include <hip/hip_runtime.h>
#include <cstddef>
#include <cstdint>

// ---------------------------------------------------------------------------
// Types for CDNA5 WMMA (wave32): v16bf = A/B fragment, v8f = C/D fragment.
// ---------------------------------------------------------------------------
typedef __attribute__((ext_vector_type(16))) __bf16        v16bf;
typedef __attribute__((ext_vector_type(8)))  float         v8f;
typedef __attribute__((ext_vector_type(4)))  unsigned int  u32x4;
typedef __attribute__((ext_vector_type(8)))  unsigned int  u32x8;
typedef __attribute__((ext_vector_type(2)))  unsigned int  u32x2;
typedef __attribute__((ext_vector_type(8)))  int           i32x8;
typedef __attribute__((ext_vector_type(4)))  int           i32x4;

#if __has_builtin(__builtin_amdgcn_tensor_load_to_lds)
#define HAS_TDM 1
#else
#define HAS_TDM 0
#endif
#if __has_builtin(__builtin_amdgcn_s_wait_tensorcnt)
#define HAS_TWAIT 1
#else
#define HAS_TWAIT 0
#endif

static __device__ __forceinline__ unsigned short f32_to_bf16(float f) {
  unsigned int u = __builtin_bit_cast(unsigned int, f);
  u += 0x7FFFu + ((u >> 16) & 1u);          // round-to-nearest-even
  return (unsigned short)(u >> 16);
}
static __device__ __forceinline__ unsigned int pack_bf16x2(float a, float b) {
  return (unsigned)f32_to_bf16(a) | ((unsigned)f32_to_bf16(b) << 16);
}

// Build a 16-element bf16 fragment from two 16-byte LDS chunks (ds_load_b128).
static __device__ __forceinline__ v16bf frag16(const unsigned short* p0,
                                               const unsigned short* p1) {
  u32x4 lo = *(const u32x4*)p0;
  u32x4 hi = *(const u32x4*)p1;
  u32x8 u;
  u[0] = lo[0]; u[1] = lo[1]; u[2] = lo[2]; u[3] = lo[3];
  u[4] = hi[0]; u[5] = hi[1]; u[6] = hi[2]; u[7] = hi[3];
  return __builtin_bit_cast(v16bf, u);
}

// Block tile: 128(M) x 128(N), K-step 32. 8 waves as 4(M)x2(N); each wave owns
// a 32x64 region = 2x4 WMMA tiles -> 8 v_wmma per wave per K-step (1.5 ds/wmma).
// Double-buffered LDS: TDM(A) + weight staging overlap with WMMA compute.
static constexpr int BM = 128;
static constexpr int BN = 128;
static constexpr int BK = 32;
static constexpr int SMEM_BYTES = 2 * (BM * BK + BN * BK) * 2;   // 32 KB

// ---------------------------------------------------------------------------
// x (fp32) -> bf16 once per call; every expert pass then streams half the bytes.
// ---------------------------------------------------------------------------
__global__ __launch_bounds__(256)
void cvt_f32_bf16x4(const float* __restrict__ in, unsigned short* __restrict__ out,
                    long n4) {
  long i = (long)blockIdx.x * 256 + threadIdx.x;
  if (i >= n4) return;
  float4 v = ((const float4*)in)[i];
  u32x2 o;
  o[0] = pack_bf16x2(v.x, v.y);
  o[1] = pack_bf16x2(v.z, v.w);
  ((u32x2*)out)[i] = o;
}

// ---------------------------------------------------------------------------
// Unified GEMM:  C = A(bf16,[M,K]) @ W(fp32,[K,N]) + bias
//  MODE 0: ACT[m][n] = bf16(gelu(C))                 (FFN layer 1)
//  MODE 1: OUT[m][n] += coef[m] * C                  (FFN layer 2, accumulate)
// A tiles via Tensor Data Mover into ping-pong LDS buffers; B tiles fp32->bf16
// transposed with loop-invariant incremental pointers.
// ---------------------------------------------------------------------------
template <int MODE>
__global__ __launch_bounds__(256)
void moe_gemm_bf16_wmma(const unsigned short* __restrict__ A,
                        const float* __restrict__ W,
                        const float* __restrict__ bias,
                        const float* __restrict__ coef, int coefStride,
                        unsigned short* __restrict__ ACT,
                        float* __restrict__ OUT,
                        int K, int N, int Mtotal) {
  extern __shared__ __align__(16) char smem[];
  unsigned short* aS = (unsigned short*)smem;   // A bufs: bytes [0, 16K)
  unsigned short* bS = aS + 2 * BM * BK;        // B bufs: bytes [16K, 32K)

  const int tid  = threadIdx.x;
  const int lane = tid & 31;
  const int wid  = tid >> 5;
  const int m0   = blockIdx.y * BM;
  const int n0   = blockIdx.x * BN;
  const int mw   = (wid & 3) * 32;    // wave M offset within block
  const int nw   = (wid >> 2) * 64;   // wave N offset within block
  const int lm   = lane & 15;
  const int sel  = (lane >> 4) & 1;   // lane-half selects K group

  // B staging: 8 fixed (nn,kp) slots per thread; advance pointers by BK*N.
  const float* wp[8];
  int bidx[8];
#pragma unroll
  for (int i = 0; i < 8; ++i) {
    int idx = tid + i * 256;
    int nn = idx & 127, kp = idx >> 7;
    wp[i]   = W + (long)(2 * kp) * N + n0 + nn;
    bidx[i] = nn * (BK / 2) + kp;
  }
  const long wstep = (long)BK * N;
  const int  nk    = K / BK;

  auto issueA = [&](int it, int buf) {
#if HAS_TDM
    if (wid == 0) {
      unsigned long long ga =
          (unsigned long long)(uintptr_t)&A[(long)m0 * K + (long)it * BK];
      u32x4 g0;
      g0[0] = 1u;                                   // count=1 (valid user D#)
      g0[1] = (unsigned)(buf * BM * BK * 2);        // lds_addr (ping-pong)
      g0[2] = (unsigned)ga;                         // global_addr[31:0]
      g0[3] = (unsigned)(ga >> 32) | (2u << 30);    // global_addr hi | type=2
      i32x8 g1;
      g1[0] = 0x00010000;                           // data_size = 2 bytes
      g1[1] = (int)(((unsigned)K & 0xFFFFu) << 16);                 // dim0 lo
      g1[2] = (int)((((unsigned)K >> 16) & 0xFFFFu) |
                    (((unsigned)Mtotal & 0xFFFFu) << 16));          // dim0 hi|dim1 lo
      g1[3] = (int)((((unsigned)Mtotal >> 16) & 0xFFFFu) |
                    ((unsigned)BK << 16));                          // dim1 hi|tile0
      g1[4] = BM;                                   // tile_dim1 (tile_dim2=0)
      g1[5] = K;                                    // tensor_dim0_stride lo32
      g1[6] = 0;
      g1[7] = 0;
      i32x4 z4 = {0, 0, 0, 0};
      i32x8 z8 = {0, 0, 0, 0, 0, 0, 0, 0};
      __builtin_amdgcn_tensor_load_to_lds(g0, g1, z4, z4, z8, 0);
    }
#else
    unsigned short* dst = aS + buf * BM * BK;
#pragma unroll
    for (int i = 0; i < 2; ++i) {
      int idx = tid + i * 256;          // 512 x 16B chunks
      int r = idx >> 2, q = idx & 3;
      ((u32x4*)dst)[idx] =
          *(const u32x4*)&A[(long)(m0 + r) * K + (long)it * BK + q * 8];
    }
#endif
  };

  auto stageB = [&](int buf) {
    unsigned int* b32 = (unsigned int*)(bS + buf * BN * BK);
#pragma unroll
    for (int i = 0; i < 8; ++i) {
      float f0 = wp[i][0];
      float f1 = wp[i][N];
      b32[bidx[i]] = pack_bf16x2(f0, f1);
      wp[i] += wstep;
    }
  };

  v8f acc[2][4] = {};

  // Prologue: stage step 0 into buffer 0.
  issueA(0, 0);
  stageB(0);

  int cur = 0;
  for (int it = 0; it < nk; ++it) {
#if HAS_TDM && HAS_TWAIT
    if (wid == 0) __builtin_amdgcn_s_wait_tensorcnt(0);  // TDM(it) done
#endif
    __syncthreads();                                     // buffers for `it` ready

    // Stage step it+1 into the alternate buffers under the WMMA shadow.
    if (it + 1 < nk) {
      issueA(it + 1, cur ^ 1);
      stageB(cur ^ 1);
      __builtin_prefetch(wp[0], 0, 1);   // warm GL2 for step it+2 weights
    }

    const unsigned short* aB = aS + cur * BM * BK;
    const unsigned short* bB = bS + cur * BN * BK;

    v16bf a[2], b[4];
#pragma unroll
    for (int ti = 0; ti < 2; ++ti) {
      const unsigned short* ar = aB + (mw + ti * 16 + lm) * BK;
      a[ti] = frag16(ar + sel * 8, ar + sel * 8 + 16);
    }
#pragma unroll
    for (int tj = 0; tj < 4; ++tj) {
      const unsigned short* br = bB + (nw + tj * 16 + lm) * BK;
      b[tj] = frag16(br + sel * 8, br + sel * 8 + 16);
    }
#pragma unroll
    for (int ti = 0; ti < 2; ++ti)
#pragma unroll
      for (int tj = 0; tj < 4; ++tj)
        acc[ti][tj] = __builtin_amdgcn_wmma_f32_16x16x32_bf16(
            false, a[ti], false, b[tj], (short)0, acc[ti][tj], false, false);
    cur ^= 1;
  }

  // ---- epilogue: C layout acc[v] -> row v + sel*8, col = lane&15 ----
#pragma unroll
  for (int ti = 0; ti < 2; ++ti) {
#pragma unroll
    for (int tj = 0; tj < 4; ++tj) {
      const int   gn = n0 + nw + tj * 16 + lm;
      const float bv = bias[gn];
#pragma unroll
      for (int v = 0; v < 8; ++v) {
        const int gm = m0 + mw + ti * 16 + v + sel * 8;
        float val = acc[ti][tj][v] + bv;
        if (MODE == 0) {
          val = 0.5f * val * (1.0f + erff(val * 0.70710678118654752f));
          ACT[(long)gm * N + gn] = f32_to_bf16(val);
        } else {
          const float cm = coef ? coef[(long)gm * coefStride] : 1.0f;
          OUT[(long)gm * N + gn] += cm * val;
        }
      }
    }
  }
}

// ---------------------------------------------------------------------------
// Router: per-token softmax(x @ gate_w + gate_b), top-k -> dense coef [NTOK,E]
// One wave32 per token.
// ---------------------------------------------------------------------------
__global__ __launch_bounds__(256)
void moe_router(const float* __restrict__ X,
                const float* __restrict__ GW,
                const float* __restrict__ GB,
                const int* __restrict__ topk_ptr,
                float* __restrict__ coef,
                int NTOK, int D, int E) {
  const int lane = threadIdx.x & 31;
  const int t    = blockIdx.x * 8 + (threadIdx.x >> 5);
  if (t >= NTOK) return;

  float acc[8];
#pragma unroll
  for (int e = 0; e < 8; ++e) acc[e] = 0.0f;

  for (int d = lane; d < D; d += 32) {
    const float xv = X[(long)t * D + d];
    const float* g = &GW[(long)d * E];
#pragma unroll
    for (int e = 0; e < 8; ++e) acc[e] += xv * g[e];
  }
#pragma unroll
  for (int e = 0; e < 8; ++e)
    for (int off = 16; off > 0; off >>= 1)
      acc[e] += __shfl_xor(acc[e], off, 32);

  float mx = -1e30f;
#pragma unroll
  for (int e = 0; e < 8; ++e) { acc[e] += GB[e]; mx = fmaxf(mx, acc[e]); }
  float p[8]; float s = 0.0f;
#pragma unroll
  for (int e = 0; e < 8; ++e) { p[e] = __expf(acc[e] - mx); s += p[e]; }
  const float inv = 1.0f / s;
#pragma unroll
  for (int e = 0; e < 8; ++e) p[e] *= inv;

  float psel[8], pw[8];
#pragma unroll
  for (int e = 0; e < 8; ++e) { psel[e] = p[e]; pw[e] = 0.0f; }
  int k = topk_ptr[0]; if (k > 8) k = 8;
  for (int kk = 0; kk < k; ++kk) {
    int j = 0; float best = psel[0];
#pragma unroll
    for (int e = 1; e < 8; ++e)
      if (psel[e] > best) { best = psel[e]; j = e; }
    pw[j] = p[j];
    psel[j] = -1.0f;
  }
  if (lane == 0) {
#pragma unroll
    for (int e = 0; e < 8; ++e) coef[(long)t * E + e] = pw[e];
  }
}

// ---------------------------------------------------------------------------
// Launch: memset out; x->bf16; router; S shared + E routed dense expert passes.
// ---------------------------------------------------------------------------
extern "C" void kernel_launch(void* const* d_in, const int* in_sizes, int n_in,
                              void* d_out, int out_size, void* d_ws, size_t ws_size,
                              hipStream_t stream) {
  const float* x    = (const float*)d_in[0];
  const float* gw   = (const float*)d_in[1];
  const float* gb   = (const float*)d_in[2];
  const float* sw1  = (const float*)d_in[3];
  const float* sb1  = (const float*)d_in[4];
  const float* sw2  = (const float*)d_in[5];
  const float* sb2  = (const float*)d_in[6];
  const float* rw1  = (const float*)d_in[7];
  const float* rb1  = (const float*)d_in[8];
  const float* rw2  = (const float*)d_in[9];
  const float* rb2  = (const float*)d_in[10];
  const int*   topk = (const int*)d_in[11];

  const int E    = in_sizes[2];               // gate_b: [E]
  const int D    = in_sizes[1] / E;           // gate_w: [D,E]
  const int H    = in_sizes[8] / E;           // rb1:    [E,H]
  const int S    = in_sizes[4] / H;           // sb1:    [S,H]
  const int NTOK = in_sizes[0] / D;           // x:      [B*T,D]

  // Workspace: xbf [NTOK,D] bf16 | act [NTOK,H] bf16 | coef [NTOK,E] fp32
  unsigned short* xbf = (unsigned short*)d_ws;
  unsigned short* act = xbf + (size_t)NTOK * D;
  float*          coef = (float*)(act + (size_t)NTOK * H);

  float* out = (float*)d_out;
  (void)hipMemsetAsync(d_out, 0, (size_t)out_size * sizeof(float), stream);

  const long n4 = (long)NTOK * D / 4;
  cvt_f32_bf16x4<<<dim3((unsigned)((n4 + 255) / 256)), dim3(256), 0, stream>>>(
      x, xbf, n4);

  moe_router<<<dim3((NTOK + 7) / 8), dim3(256), 0, stream>>>(
      x, gw, gb, topk, coef, NTOK, D, E);

  const dim3 blk(256);
  const dim3 g1(H / BN, NTOK / BM);   // layer-1 GEMM: N = H
  const dim3 g2(D / BN, NTOK / BM);   // layer-2 GEMM: N = D

  for (int s = 0; s < S; ++s) {
    moe_gemm_bf16_wmma<0><<<g1, blk, SMEM_BYTES, stream>>>(
        xbf, sw1 + (size_t)s * D * H, sb1 + (size_t)s * H,
        nullptr, 0, act, nullptr, D, H, NTOK);
    moe_gemm_bf16_wmma<1><<<g2, blk, SMEM_BYTES, stream>>>(
        act, sw2 + (size_t)s * H * D, sb2 + (size_t)s * D,
        nullptr, 0, nullptr, out, H, D, NTOK);
  }
  for (int e = 0; e < E; ++e) {
    moe_gemm_bf16_wmma<0><<<g1, blk, SMEM_BYTES, stream>>>(
        xbf, rw1 + (size_t)e * D * H, rb1 + (size_t)e * H,
        nullptr, 0, act, nullptr, D, H, NTOK);
    moe_gemm_bf16_wmma<1><<<g2, blk, SMEM_BYTES, stream>>>(
        act, rw2 + (size_t)e * H * D, rb2 + (size_t)e * D,
        coef + e, E, nullptr, out, H, D, NTOK);
  }
}
